// LearnedGCN_455266533808
// MI455X (gfx1250) — matrix operand order repaired and checked by reading, not compiled, not run on hardware
//
#include <hip/hip_runtime.h>

typedef __attribute__((ext_vector_type(2)))  float    v2f;
typedef __attribute__((ext_vector_type(4)))  float    v4f;
typedef __attribute__((ext_vector_type(8)))  float    v8f;
typedef __attribute__((ext_vector_type(16))) _Float16 v16h;

#define GN    16384
#define GRANK 10
#define GF    64
#define NTILE (GN / 16)              // 1024 column tiles
#define KTP_FLOATS (NTILE * 3 * 64)  // tile x chunk x (32 lanes x v2f)
#define L2E   1.4426950408889634f

// Prologue 1: x [N,64] f32  ->  xt [64,N] f16  (transposed + converted)
__global__ __launch_bounds__(256) void xt_kernel(const float* __restrict__ x,
                                                 _Float16* __restrict__ xt) {
  int t = blockIdx.x * 256 + threadIdx.x;   // 0 .. 64*16384-1
  int f = t >> 14;
  int j = t & (GN - 1);
  xt[f * GN + j] = (_Float16)x[j * GF + f];
}

// Prologue 2: target_embed [10,N] -> ktp, pre-swizzled into the f32-WMMA
// A-operand layout, zero-padded RANK 10->12, so each A operand in the hot
// loop is one coalesced b64 load.
__global__ __launch_bounds__(256) void ktp_kernel(const float* __restrict__ te,
                                                  float* __restrict__ ktp) {
  int t = blockIdx.x * 256 + threadIdx.x;   // 0 .. KTP_FLOATS-1
  int v    = t & 1;
  int lane = (t >> 1) & 31;
  int cc   = t >> 6;
  int c    = cc % 3;
  int tile = cc / 3;
  int k = 4 * c + 2 * (lane >> 4) + v;
  int j = tile * 16 + (lane & 15);
  ktp[t] = (k < GRANK) ? te[k * GN + j] : 0.0f;
}

// Fused: out = softmax(relu(Q @ Kt), row) @ x @ W^T + b
// One wave per 16-row tile; scores computed transposed so row index lives in
// lane%16 (reductions are per-lane + one xor-16 shuffle) and the exp'd tiles
// map 1:1 onto the f16 WMMA A-operand layout (no cross-lane movement).
__global__ __launch_bounds__(256) void gcn_kernel(
    const float*    __restrict__ Q,    // source_embed [N, RANK]
    const float*    __restrict__ ktp,  // pre-swizzled target_embed
    const _Float16* __restrict__ xt,   // x^T as f16 [F, N]
    const float*    __restrict__ W,    // [F, F]
    const float*    __restrict__ bias, // [F]
    float*          __restrict__ out)  // [N, F]
{
  __shared__ __align__(16) float lds[8 * 16 * 68];
  const int lane = threadIdx.x & 31;
  const int wid  = threadIdx.x >> 5;
  const int half = lane >> 4;
  const int l16  = lane & 15;
  const int i0   = (blockIdx.x * 8 + wid) * 16;

  const v8f vz = {0.f, 0.f, 0.f, 0.f, 0.f, 0.f, 0.f, 0.f};

  // Resident B operand (f32 16x16x4): B(kk,n) = Q[i0+n][k0+kk]; once per wave.
  v2f bq[3];
  #pragma unroll
  for (int c = 0; c < 3; ++c) {
    #pragma unroll
    for (int v = 0; v < 2; ++v) {
      int k = 4 * c + v + 2 * half;
      bq[c][v] = (k < GRANK) ? Q[(i0 + l16) * GRANK + k] : 0.0f;
    }
  }

  const v2f* ka = (const v2f*)ktp + lane;   // + tile*96 + c*32 per operand
  const _Float16* xr = xt + l16 * GN + 8 * half;

  // ---------------- Pass A: exact row max of scores ----------------
  // 8 independent running maxes -> no serial VALU chain inside the loop.
  float m8[8];
  #pragma unroll
  for (int v = 0; v < 8; ++v) m8[v] = -1e30f;

  for (int t = 0; t < NTILE; t += 2) {
    const v2f* k0 = ka + t * 96;
    v8f d0 = __builtin_amdgcn_wmma_f32_16x16x4_f32(false, k0[0],   false, bq[0], (short)0, vz, false, false);
    v8f d1 = __builtin_amdgcn_wmma_f32_16x16x4_f32(false, k0[96],  false, bq[0], (short)0, vz, false, false);
    d0     = __builtin_amdgcn_wmma_f32_16x16x4_f32(false, k0[32],  false, bq[1], (short)0, d0, false, false);
    d1     = __builtin_amdgcn_wmma_f32_16x16x4_f32(false, k0[128], false, bq[1], (short)0, d1, false, false);
    d0     = __builtin_amdgcn_wmma_f32_16x16x4_f32(false, k0[64],  false, bq[2], (short)0, d0, false, false);
    d1     = __builtin_amdgcn_wmma_f32_16x16x4_f32(false, k0[160], false, bq[2], (short)0, d1, false, false);
    #pragma unroll
    for (int v = 0; v < 8; ++v) m8[v] = fmaxf(m8[v], fmaxf(d0[v], d1[v]));  // v_max3
  }
  float m = fmaxf(fmaxf(fmaxf(m8[0], m8[1]), fmaxf(m8[2], m8[3])),
                  fmaxf(fmaxf(m8[4], m8[5]), fmaxf(m8[6], m8[7])));
  m = fmaxf(m, __shfl_xor(m, 16, 32));  // lanes L, L+16 hold same row i = l16
  m = fmaxf(m, 0.0f);                   // relu floor
  const float mneg = -m * L2E;          // exp(relu(s)-m) = exp2(fma(relu(s),L2E,mneg))

  // ---------------- Pass B: p = exp(relu(s)-m); acc += P @ V ----------------
  v8f acc[4];
  #pragma unroll
  for (int fc = 0; fc < 4; ++fc) acc[fc] = vz;
  float ls[8];
  #pragma unroll
  for (int v = 0; v < 8; ++v) ls[v] = 0.0f;

  for (int t = 0; t < NTILE; t += 2) {
    const v2f* k0 = ka + t * 96;
    v8f d0 = __builtin_amdgcn_wmma_f32_16x16x4_f32(false, k0[0],   false, bq[0], (short)0, vz, false, false);
    v8f d1 = __builtin_amdgcn_wmma_f32_16x16x4_f32(false, k0[96],  false, bq[0], (short)0, vz, false, false);
    d0     = __builtin_amdgcn_wmma_f32_16x16x4_f32(false, k0[32],  false, bq[1], (short)0, d0, false, false);
    d1     = __builtin_amdgcn_wmma_f32_16x16x4_f32(false, k0[128], false, bq[1], (short)0, d1, false, false);
    d0     = __builtin_amdgcn_wmma_f32_16x16x4_f32(false, k0[64],  false, bq[2], (short)0, d0, false, false);
    d1     = __builtin_amdgcn_wmma_f32_16x16x4_f32(false, k0[160], false, bq[2], (short)0, d1, false, false);

    // relu + exp2 + rowsum; element v of the transposed-score tile is exactly
    // half v (tile0) / v+8 (tile1) of the f16 A operand.
    v16h p;
    #pragma unroll
    for (int v = 0; v < 8; ++v) {
      float e0 = __builtin_amdgcn_exp2f(fmaf(fmaxf(d0[v], 0.0f), L2E, mneg));
      float e1 = __builtin_amdgcn_exp2f(fmaf(fmaxf(d1[v], 0.0f), L2E, mneg));
      ls[v] += e0 + e1;
      p[v]     = (_Float16)e0;
      p[v + 8] = (_Float16)e1;
    }

    // B operand (32x16 f16): lane n = l16 (feature), K = 8*half + 0..7 | 16 + 8*half + 0..7
    #pragma unroll
    for (int fc = 0; fc < 4; ++fc) {
      const _Float16* xp = xr + fc * (16 * GN) + t * 16;
      v16h bv;
      #pragma unroll
      for (int s = 0; s < 8; ++s) { bv[s] = xp[s]; bv[s + 8] = xp[16 + s]; }
      acc[fc] = __builtin_amdgcn_wmma_f32_16x16x32_f16(false, p, false, bv,
                                                       (short)0, acc[fc], false, false);
    }
  }

  float lsum = ((ls[0] + ls[1]) + (ls[2] + ls[3])) + ((ls[4] + ls[5]) + (ls[6] + ls[7]));
  lsum += __shfl_xor(lsum, 16, 32);   // full row sum, aligned to row i = l16
  float rl = 1.0f / lsum;

  // Stage unnormalized agg tile to LDS (row = v+8*half, col = fc*16+l16; pad 68)
  float* myl = lds + wid * (16 * 68);
  #pragma unroll
  for (int fc = 0; fc < 4; ++fc)
    #pragma unroll
    for (int v = 0; v < 8; ++v)
      myl[(v + 8 * half) * 68 + fc * 16 + l16] = acc[fc][v];

  __syncthreads();

  // Final linear: out[i0+i, o] = rl * sum_f agg[i,f] * W[o,f] + b[o]
  // t = lane + 32*iter -> i = t&15 == l16, so rl is already row-aligned.
  for (int t = lane; t < 16 * GF; t += 32) {
    int i = t & 15;
    int o = t >> 4;
    const v4f* ar = (const v4f*)(myl + i * 68);   // i*68 floats = 16B-aligned
    const v4f* wr = (const v4f*)(W + o * GF);
    v4f s4 = {0.f, 0.f, 0.f, 0.f};
    #pragma unroll
    for (int fq = 0; fq < GF / 4; ++fq) s4 += ar[fq] * wr[fq];
    float s = s4[0] + s4[1] + s4[2] + s4[3];
    out[(i0 + i) * GF + o] = rl * s + bias[o];
  }
}

extern "C" void kernel_launch(void* const* d_in, const int* in_sizes, int n_in,
                              void* d_out, int out_size, void* d_ws, size_t ws_size,
                              hipStream_t stream) {
  const float* x  = (const float*)d_in[0];  // [N, 64]
  const float* se = (const float*)d_in[1];  // [N, 10]
  const float* te = (const float*)d_in[2];  // [10, N]
  const float* W  = (const float*)d_in[3];  // [64, 64]
  const float* b  = (const float*)d_in[4];  // [64]
  float* out = (float*)d_out;               // [N, 64]

  _Float16* xt = (_Float16*)d_ws;                              // 2 MB
  float* ktp   = (float*)((char*)d_ws + (size_t)GF * GN * 2);  // 768 KB

  xt_kernel<<<(GF * GN) / 256, 256, 0, stream>>>(x, xt);
  ktp_kernel<<<KTP_FLOATS / 256, 256, 0, stream>>>(te, ktp);
  gcn_kernel<<<GN / 128, 256, 0, stream>>>(se, ktp, xt, W, b, out);
}
